// GCN_Actor_24223615550523
// MI455X (gfx1250) — matrix operand order repaired and checked by reading, not compiled, not run on hardware
//
#include <hip/hip_runtime.h>
#include <hip/hip_bf16.h>

#define BB      8
#define NET_N   8192
#define DAG_N   256
#define DIM     128
#define F_NET   64
#define F_DAG   32
#define E_NET   1048576
#define E_DAG   16384

typedef __attribute__((ext_vector_type(16))) __bf16 v16bf;
typedef __attribute__((ext_vector_type(8)))  float  v8f;

__device__ __forceinline__ unsigned short f32_to_bf16_rne(float f) {
  unsigned int u = __float_as_uint(f);
  unsigned int r = u + 0x7FFFu + ((u >> 16) & 1u);
  return (unsigned short)(r >> 16);
}
__device__ __forceinline__ float bf16u_to_f32(unsigned short h) {
  return __uint_as_float(((unsigned int)h) << 16);
}

// Pack W[K x 128] into fragment-ready bf16 hi/lo layout:
// for (kstep s, ntile, lane): 16 hi ushorts then 16 lo ushorts, contiguous 64B.
// Flat index: (((s*8 + ntile)*32 + lane)*32 + j)  ==  t*32 + j with t = s*256 + ntile*32 + lane.
__global__ __launch_bounds__(256)
void pack_w_kernel(const float* __restrict__ W, unsigned short* __restrict__ Wpk, int K) {
  const int t = blockIdx.x * blockDim.x + threadIdx.x;   // grid = (K/32) blocks of 256
  const int r = t & 255;
  const int s = t >> 8;                                  // kstep
  const int ntile = r >> 5;
  const int lane = r & 31;
  const int half = lane >> 4;
  const int ncol = ntile * 16 + (lane & 15);
  unsigned short* dst = Wpk + (size_t)t * 32;
#pragma unroll
  for (int j = 0; j < 16; ++j) {
    const float wv = W[(size_t)(s * 32 + 16 * half + j) * DIM + ncol];
    const unsigned short hb = f32_to_bf16_rne(wv);
    dst[j] = hb;
    dst[16 + j] = f32_to_bf16_rne(wv - bf16u_to_f32(hb));
  }
}

// ---------------- WMMA GEMM: Out[M x 128] = (relu?)X[M x K] @ W[K x 128] ----------------
// Block = 8 waves; block handles one 16-row tile, wave w handles n-tile w.
// A fragments converted once per block into LDS (hi/lo bf16); B prepacked in global.
// f32 accuracy via bf16 split: x*w ~= xh*wh + xl*wh + xh*wl.
__global__ __launch_bounds__(256)
void gemm_xw_wmma(const float* __restrict__ X, const unsigned short* __restrict__ Wpk,
                  float* __restrict__ Out, int M, int K, int applyRelu) {
  __shared__ unsigned short sA[4096];        // up to (K/32=4) ksteps * 32 lanes * 32 ushorts = 8KB
  const int m0 = blockIdx.x * 16;

  // Phase 1: convert A tile into fragment layout (thread t handles (kstep, lane) pair).
  const int t = threadIdx.x;
  if (t < K) {                               // K/32 ksteps * 32 lanes == K pairs
    const int s = t >> 5;
    const int l = t & 31;
    const int half = l >> 4;
    const int mrow = m0 + (l & 15);
    const float* xr = X + (size_t)mrow * K + s * 32 + 16 * half;
    const float4 a0 = *(const float4*)(xr + 0);    // k = s*32+16h + 0..3
    const float4 a1 = *(const float4*)(xr + 4);    // .. +4..7
    const float4 a2 = *(const float4*)(xr + 8);    // .. +8..11
    const float4 a3 = *(const float4*)(xr + 12);   // .. +12..15
    float av[16] = { a0.x, a0.y, a0.z, a0.w, a1.x, a1.y, a1.z, a1.w,
                     a2.x, a2.y, a2.z, a2.w, a3.x, a3.y, a3.z, a3.w };
    unsigned short* dst = sA + (size_t)t * 32;
#pragma unroll
    for (int j = 0; j < 16; ++j) {
      float v = av[j];
      if (applyRelu) v = fmaxf(v, 0.0f);
      const unsigned short hb = f32_to_bf16_rne(v);
      dst[j] = hb;
      dst[16 + j] = f32_to_bf16_rne(v - bf16u_to_f32(hb));
    }
  }
  __syncthreads();

  // Phase 2: each wave accumulates its 16x16 tile.
  const int lane = threadIdx.x & 31;
  const int wave = threadIdx.x >> 5;
  const int half = lane >> 4;
  const int l15 = lane & 15;
  v8f acc = {};
  const int ksteps = K >> 5;
  for (int s = 0; s < ksteps; ++s) {
    const unsigned short* ap = sA + ((size_t)(s * 32 + lane)) * 32;
    const v16bf a_h = *(const v16bf*)(ap);
    const v16bf a_l = *(const v16bf*)(ap + 16);
    const unsigned short* bp = Wpk + ((size_t)((s * 8 + wave) * 32 + lane)) * 32;
    const v16bf b_h = *(const v16bf*)(bp);
    const v16bf b_l = *(const v16bf*)(bp + 16);
    acc = __builtin_amdgcn_wmma_f32_16x16x32_bf16(false, a_h, false, b_h,
                                                  (short)0, acc, false, false);
    acc = __builtin_amdgcn_wmma_f32_16x16x32_bf16(false, a_l, false, b_h,
                                                  (short)0, acc, false, false);
    acc = __builtin_amdgcn_wmma_f32_16x16x32_bf16(false, a_h, false, b_l,
                                                  (short)0, acc, false, false);
  }
#pragma unroll
  for (int r = 0; r < 8; ++r) {
    const int m = m0 + r + 8 * half;         // D layout: VGPR r -> rows r / r+8
    Out[(size_t)m * DIM + wave * 16 + l15] = acc[r];
  }
}

// ---------------- small utility kernels ----------------
__global__ void fill_kernel(float* p, float v, int n) {
  int i = blockIdx.x * blockDim.x + threadIdx.x;
  if (i < n) p[i] = v;
}
__global__ void deg_accum_kernel(const int* __restrict__ ei, const float* __restrict__ ew,
                                 float* __restrict__ deg, int E) {
  int e = blockIdx.x * blockDim.x + threadIdx.x;
  if (e < E) atomicAdd(&deg[ei[E + e]], ew[e]);   // col = edge_index[1][e]
}
__global__ void deg_to_dinv_kernel(float* deg, int n) {
  int i = blockIdx.x * blockDim.x + threadIdx.x;
  if (i < n) { float d = deg[i]; deg[i] = (d > 0.0f) ? rsqrtf(d) : 0.0f; }
}
// out[i,f] = dinv[i]^2 * xw[i,f] + bias[f]   (self-loop term + bias)
__global__ void init_out_kernel(const float* __restrict__ xw, const float* __restrict__ dinv,
                                const float* __restrict__ bias, float* __restrict__ out, int n) {
  int idx = blockIdx.x * blockDim.x + threadIdx.x;
  if (idx < n) {
    int i = idx >> 7, f = idx & 127;
    float di = dinv[i];
    out[idx] = di * di * xw[idx] + bias[f];
  }
}
// One wave per edge; lane handles 4 features.
__global__ __launch_bounds__(256)
void edge_scatter_kernel(const int* __restrict__ ei, const float* __restrict__ ew,
                         const float* __restrict__ dinv, const float* __restrict__ xw,
                         float* __restrict__ out, int E) {
  const int lane = threadIdx.x & 31;
  const int nw = (gridDim.x * blockDim.x) >> 5;
  for (int e = (blockIdx.x * blockDim.x + threadIdx.x) >> 5; e < E; e += nw) {
    const int row = ei[e];
    const int col = ei[E + e];
    const float norm = dinv[row] * ew[e] * dinv[col];
    const float4 x4 = *(const float4*)(xw + (size_t)row * DIM + lane * 4);
    float* op = out + (size_t)col * DIM + lane * 4;
    atomicAdd(op + 0, norm * x4.x);
    atomicAdd(op + 1, norm * x4.y);
    atomicAdd(op + 2, norm * x4.z);
    atomicAdd(op + 3, norm * x4.w);
  }
}
// Fused: relu -> l2norm (in place) -> nodedot = x . wvec -> batch-mean accumulation.
__global__ __launch_bounds__(256)
void finalize_embed_kernel(float* __restrict__ x, const float* __restrict__ wvec,
                           float* __restrict__ nodedot, float* __restrict__ meanacc,
                           int M, int nodesPerBatch) {
  const int wid = (blockIdx.x * blockDim.x + threadIdx.x) >> 5;
  const int lane = threadIdx.x & 31;
  if (wid >= M) return;
  float4 v = *(const float4*)(x + (size_t)wid * DIM + lane * 4);
  v.x = fmaxf(v.x, 0.f); v.y = fmaxf(v.y, 0.f); v.z = fmaxf(v.z, 0.f); v.w = fmaxf(v.w, 0.f);
  float ss = v.x * v.x + v.y * v.y + v.z * v.z + v.w * v.w;
#pragma unroll
  for (int m = 16; m >= 1; m >>= 1) ss += __shfl_xor(ss, m, 32);
  const float inv = 1.0f / fmaxf(sqrtf(ss), 1e-12f);
  v.x *= inv; v.y *= inv; v.z *= inv; v.w *= inv;
  *(float4*)(x + (size_t)wid * DIM + lane * 4) = v;
  const float4 w4 = *(const float4*)(wvec + lane * 4);
  float d = v.x * w4.x + v.y * w4.y + v.z * w4.z + v.w * w4.w;
#pragma unroll
  for (int m = 16; m >= 1; m >>= 1) d += __shfl_xor(d, m, 32);
  if (lane == 0) nodedot[wid] = d;
  const int b = wid / nodesPerBatch;
  const float sc = 1.0f / (float)nodesPerBatch;
  atomicAdd(&meanacc[b * DIM + lane * 4 + 0], v.x * sc);
  atomicAdd(&meanacc[b * DIM + lane * 4 + 1], v.y * sc);
  atomicAdd(&meanacc[b * DIM + lane * 4 + 2], v.z * sc);
  atomicAdd(&meanacc[b * DIM + lane * 4 + 3], v.w * sc);
}
// abar[b,j] = mean_i softmax_j( leakyrelu(ni[b,i] + dj[b,j] + 1e-7) )
__global__ __launch_bounds__(256)
void attn_abar_kernel(const float* __restrict__ ni, const float* __restrict__ dj,
                      float* __restrict__ abar, int chunks) {
  __shared__ float sdj[DAG_N];
  __shared__ float sacc[DAG_N];
  const int b = blockIdx.x / chunks;
  const int chunk = blockIdx.x % chunks;
  sdj[threadIdx.x] = dj[b * DAG_N + threadIdx.x];
  sacc[threadIdx.x] = 0.0f;
  __syncthreads();
  const int lane = threadIdx.x & 31;
  const int wv = threadIdx.x >> 5;
  const int rows = NET_N / chunks;
  for (int r = wv; r < rows; r += 8) {
    const float niv = ni[b * NET_N + chunk * rows + r];
    float ev[8];
    float mx = -3.0e38f;
#pragma unroll
    for (int t = 0; t < 8; ++t) {
      float e = niv + sdj[lane + 32 * t] + 1e-7f;
      e = (e >= 0.0f) ? e : 0.01f * e;
      ev[t] = e;
      mx = fmaxf(mx, e);
    }
#pragma unroll
    for (int m = 16; m >= 1; m >>= 1) mx = fmaxf(mx, __shfl_xor(mx, m, 32));
    float s = 0.0f;
#pragma unroll
    for (int t = 0; t < 8; ++t) { ev[t] = expf(ev[t] - mx); s += ev[t]; }
#pragma unroll
    for (int m = 16; m >= 1; m >>= 1) s += __shfl_xor(s, m, 32);
    const float invs = 1.0f / s;
#pragma unroll
    for (int t = 0; t < 8; ++t) atomicAdd(&sacc[lane + 32 * t], ev[t] * invs);
  }
  __syncthreads();
  atomicAdd(&abar[b * DAG_N + threadIdx.x], sacc[threadIdx.x] * (1.0f / (float)NET_N));
}
// hyb = 0.5 * ( net_mean*(1 + NET_N/DAG_N) + dag_mean + abar @ dag_e )
__global__ void compute_hyb_kernel(const float* __restrict__ net_mean,
                                   const float* __restrict__ dag_mean,
                                   const float* __restrict__ abar,
                                   const float* __restrict__ dag_e,
                                   float* __restrict__ hyb) {
  const int b = blockIdx.x;
  const int d = threadIdx.x;                 // 128 threads
  float s = 0.0f;
  for (int j = 0; j < DAG_N; ++j)
    s += abar[b * DAG_N + j] * dag_e[((size_t)b * DAG_N + j) * DIM + d];
  const float nm = net_mean[b * DIM + d];
  const float dm = dag_mean[b * DIM + d];
  hyb[b * DIM + d] = 0.5f * (nm * (1.0f + (float)NET_N / (float)DAG_N) + dm + s);
}
__global__ void head1_kernel(const float* __restrict__ hyb, const float* __restrict__ W1,
                             const float* __restrict__ b1, float* __restrict__ h) {
  const int b = blockIdx.x, o = threadIdx.x; // 256 threads
  float s = b1[o];
  for (int k = 0; k < DIM; ++k) s += hyb[b * DIM + k] * W1[k * 256 + o];
  h[b * 256 + o] = fmaxf(s, 0.0f);
}
__global__ void head2_kernel(const float* __restrict__ h, const float* __restrict__ W2,
                             const float* __restrict__ b2, float* __restrict__ out) {
  const int idx = blockIdx.x * blockDim.x + threadIdx.x; // B*2048
  const int b = idx >> 11, o = idx & 2047;
  float s = b2[o];
  for (int k = 0; k < 256; ++k) s += h[b * 256 + k] * W2[k * 2048 + o];
  out[idx] = s;
}

// ---------------- launch ----------------
extern "C" void kernel_launch(void* const* d_in, const int* in_sizes, int n_in,
                              void* d_out, int out_size, void* d_ws, size_t ws_size,
                              hipStream_t stream) {
  const float* net_feat = (const float*)d_in[0];
  const int*   net_ei   = (const int*)  d_in[1];
  const float* net_ew   = (const float*)d_in[2];
  const float* dag_feat = (const float*)d_in[3];
  const int*   dag_ei   = (const int*)  d_in[4];
  const float* dag_ew   = (const float*)d_in[5];
  const float* net_W0   = (const float*)d_in[8];
  const float* net_b0   = (const float*)d_in[9];
  const float* net_W1   = (const float*)d_in[10];
  const float* net_b1   = (const float*)d_in[11];
  const float* dag_W0   = (const float*)d_in[12];
  const float* dag_b0   = (const float*)d_in[13];
  const float* dag_W1   = (const float*)d_in[14];
  const float* dag_b1   = (const float*)d_in[15];
  const float* W_a      = (const float*)d_in[16];
  const float* head_W1  = (const float*)d_in[17];
  const float* head_b1  = (const float*)d_in[18];
  const float* head_W2  = (const float*)d_in[19];
  const float* head_b2  = (const float*)d_in[20];

  const int MN = BB * NET_N;   // 65536
  const int MD = BB * DAG_N;   // 2048

  float* ws = (float*)d_ws;
  size_t off = 0;
  float* dinv_net = ws + off; off += MN;
  float* dinv_dag = ws + off; off += MD;
  float* bufA     = ws + off; off += (size_t)MN * DIM;
  float* bufB     = ws + off; off += (size_t)MN * DIM;
  float* dagA     = ws + off; off += (size_t)MD * DIM;
  float* dagB     = ws + off; off += (size_t)MD * DIM;
  float* ni       = ws + off; off += MN;
  float* dj       = ws + off; off += MD;
  float* net_mean = ws + off; off += BB * DIM;   // next three contiguous (zeroed together)
  float* dag_mean = ws + off; off += BB * DIM;
  float* abar     = ws + off; off += BB * DAG_N;
  float* hyb      = ws + off; off += BB * DIM;
  float* h        = ws + off; off += BB * 256;
  off = (off + 15) & ~(size_t)15;                // 64B align for packed weights
  unsigned short* wpk = (unsigned short*)(ws + off); off += 16384;  // 32768 ushorts (K<=128)

  // zero accumulators (net_mean | dag_mean | abar contiguous = 4096 floats)
  fill_kernel<<<16, 256, 0, stream>>>(net_mean, 0.0f, 4096);

  // ---- degrees / dinv ----
  fill_kernel<<<MN / 256, 256, 0, stream>>>(dinv_net, 1.0f, MN);   // self-loop weight 1
  deg_accum_kernel<<<E_NET / 256, 256, 0, stream>>>(net_ei, net_ew, dinv_net, E_NET);
  deg_to_dinv_kernel<<<MN / 256, 256, 0, stream>>>(dinv_net, MN);
  fill_kernel<<<MD / 256, 256, 0, stream>>>(dinv_dag, 1.0f, MD);
  deg_accum_kernel<<<E_DAG / 256, 256, 0, stream>>>(dag_ei, dag_ew, dinv_dag, E_DAG);
  deg_to_dinv_kernel<<<MD / 256, 256, 0, stream>>>(dinv_dag, MD);

  const int nElemN = MN * DIM, nElemD = MD * DIM;

  // ---- net GCN layer 0 ----
  pack_w_kernel<<<F_NET / 32, 256, 0, stream>>>(net_W0, wpk, F_NET);
  gemm_xw_wmma<<<MN / 16, 256, 0, stream>>>(net_feat, wpk, bufA, MN, F_NET, 0);
  init_out_kernel<<<nElemN / 256, 256, 0, stream>>>(bufA, dinv_net, net_b0, bufB, nElemN);
  edge_scatter_kernel<<<E_NET / 8, 256, 0, stream>>>(net_ei, net_ew, dinv_net, bufA, bufB, E_NET);
  // ---- net GCN layer 1 (relu folded into GEMM A-conversion) ----
  pack_w_kernel<<<DIM / 32, 256, 0, stream>>>(net_W1, wpk, DIM);
  gemm_xw_wmma<<<MN / 16, 256, 0, stream>>>(bufB, wpk, bufA, MN, DIM, 1);
  init_out_kernel<<<nElemN / 256, 256, 0, stream>>>(bufA, dinv_net, net_b1, bufB, nElemN);
  edge_scatter_kernel<<<E_NET / 8, 256, 0, stream>>>(net_ei, net_ew, dinv_net, bufA, bufB, E_NET);
  // relu + l2norm + ni = net_e.wn + net_mean accumulation
  finalize_embed_kernel<<<MN / 8, 256, 0, stream>>>(bufB, W_a, ni, net_mean, MN, NET_N);

  // ---- dag GCN layer 0 ----
  pack_w_kernel<<<F_DAG / 32, 256, 0, stream>>>(dag_W0, wpk, F_DAG);
  gemm_xw_wmma<<<MD / 16, 256, 0, stream>>>(dag_feat, wpk, dagA, MD, F_DAG, 0);
  init_out_kernel<<<nElemD / 256, 256, 0, stream>>>(dagA, dinv_dag, dag_b0, dagB, nElemD);
  edge_scatter_kernel<<<E_DAG / 8, 256, 0, stream>>>(dag_ei, dag_ew, dinv_dag, dagA, dagB, E_DAG);
  // ---- dag GCN layer 1 (relu folded into GEMM) ----
  pack_w_kernel<<<DIM / 32, 256, 0, stream>>>(dag_W1, wpk, DIM);
  gemm_xw_wmma<<<MD / 16, 256, 0, stream>>>(dagB, wpk, dagA, MD, DIM, 1);
  init_out_kernel<<<nElemD / 256, 256, 0, stream>>>(dagA, dinv_dag, dag_b1, dagB, nElemD);
  edge_scatter_kernel<<<E_DAG / 8, 256, 0, stream>>>(dag_ei, dag_ew, dinv_dag, dagA, dagB, E_DAG);
  finalize_embed_kernel<<<MD / 8, 256, 0, stream>>>(dagB, W_a + DIM, dj, dag_mean, MD, DAG_N);

  // ---- attention (collapsed to softmax row-mean; softmax rows sum to 1) ----
  const int chunks = 64;
  attn_abar_kernel<<<BB * chunks, 256, 0, stream>>>(ni, dj, abar, chunks);

  // ---- hyb + head ----
  compute_hyb_kernel<<<BB, DIM, 0, stream>>>(net_mean, dag_mean, abar, dagB, hyb);
  head1_kernel<<<BB, 256, 0, stream>>>(hyb, head_W1, head_b1, h);
  head2_kernel<<<(BB * 2048) / 256, 256, 0, stream>>>(h, head_W2, head_b2, (float*)d_out);
}
